// Backbone_44787918963380
// MI455X (gfx1250) — matrix operand order repaired and checked by reading, not compiled, not run on hardware
//
#include <hip/hip_runtime.h>

// ---------------------------------------------------------------------------
// MI455X / gfx1250 graph-attention backbone.
// Dense GEMMs run through V_WMMA_F32_16X16X32_F16:
//   - weights pre-transposed + f16 ([N x Kpad], zero-padded so K%32==0 always)
//   - 16x64 tile per wave, 4 WMMAs per A-fragment
//   - two load paths only: full-tile vector / row-masked M-tail vector
// Softmax aggregation: ordered-uint atomicMax + float atomicAdd, vectorized.
// ---------------------------------------------------------------------------

typedef _Float16 half_t;
typedef __attribute__((ext_vector_type(16))) _Float16 v16h;
typedef __attribute__((ext_vector_type(8)))  _Float16 v8h;
typedef __attribute__((ext_vector_type(4)))  _Float16 v4h;
typedef __attribute__((ext_vector_type(8)))  float    v8f;

#define KN 8192   // total k-nodes  (A * T)
#define AN 512    // agents
#define TN 16     // intervals
#define HN 128    // hidden
#define GN 3072   // polygons
#define FPAD 32   // padded edge-feature width (K of first edge GEMM)

// ---- fragment loaders (8 consecutive K-elements into af[off..off+7]) ------
__device__ __forceinline__ void loadA8(v16h& f, int off, const float* p) {
  float4 x = *(const float4*)p;
  float4 y = *(const float4*)(p + 4);
  f[off + 0] = (half_t)x.x; f[off + 1] = (half_t)x.y;
  f[off + 2] = (half_t)x.z; f[off + 3] = (half_t)x.w;
  f[off + 4] = (half_t)y.x; f[off + 5] = (half_t)y.y;
  f[off + 6] = (half_t)y.z; f[off + 7] = (half_t)y.w;
}
__device__ __forceinline__ void loadA8(v16h& f, int off, const half_t* p) {
  v8h v = *(const v8h*)p;
#pragma unroll
  for (int i = 0; i < 8; ++i) f[off + i] = v[i];
}

// ---------------------------------------------------------------------------
// WMMA GEMM:  C[M x N] = A[M x K] @ W[K x N] + bias (+residual) (relu)
// Wt = transposed f16 weights [N x K].  Requires K % 32 == 0 (zero-padded).
// One 16x64 tile per wave, 4 waves per block.
// ---------------------------------------------------------------------------
template <typename TA, typename TC>
__global__ __launch_bounds__(128)
void wmma_gemm_kernel(const TA* __restrict__ A, int lda,
                      const half_t* __restrict__ Wt,
                      const float* __restrict__ bias,
                      const float* __restrict__ residual,
                      TC* __restrict__ C, int ldc,
                      int M, int N, int Kdim, int relu)
{
  const int wave  = threadIdx.x >> 5;
  const int lane  = threadIdx.x & 31;
  const int tileM = blockIdx.x * 4 + wave;
  if (tileM * 16 >= M) return;             // wave-uniform exit
  const int half  = lane >> 4;
  const int mr    = lane & 15;
  const int row   = tileM * 16 + mr;
  const int ncol0 = blockIdx.y * 64;

  v8f acc[4] = {};
  const half_t* w0   = Wt + (long)(ncol0 + mr) * Kdim;
  const TA*     arow = A + (long)row * lda;

  if ((tileM * 16 + 16) <= M) {
    // ---- full tile: all loads vector + unguarded --------------------------
    for (int k0 = 0; k0 < Kdim; k0 += 32) {
      v16h af;
      loadA8(af, 0, arow + k0 + half * 16);
      loadA8(af, 8, arow + k0 + half * 16 + 8);
      const int wofs = k0 + half * 16;
      v16h b0 = *(const v16h*)(w0 + wofs);
      v16h b1 = *(const v16h*)(w0 + (long)16 * Kdim + wofs);
      v16h b2 = *(const v16h*)(w0 + (long)32 * Kdim + wofs);
      v16h b3 = *(const v16h*)(w0 + (long)48 * Kdim + wofs);
      acc[0] = __builtin_amdgcn_wmma_f32_16x16x32_f16(false, af, false, b0, (short)0, acc[0], false, false);
      acc[1] = __builtin_amdgcn_wmma_f32_16x16x32_f16(false, af, false, b1, (short)0, acc[1], false, false);
      acc[2] = __builtin_amdgcn_wmma_f32_16x16x32_f16(false, af, false, b2, (short)0, acc[2], false, false);
      acc[3] = __builtin_amdgcn_wmma_f32_16x16x32_f16(false, af, false, b3, (short)0, acc[3], false, false);
    }
  } else {
    // ---- M-tail tile: mask rows once per fragment, B unguarded ------------
    const bool rowOK = row < M;
    for (int k0 = 0; k0 < Kdim; k0 += 32) {
      v16h af = {};
      if (rowOK) {
        loadA8(af, 0, arow + k0 + half * 16);
        loadA8(af, 8, arow + k0 + half * 16 + 8);
      }
      const int wofs = k0 + half * 16;
      v16h b0 = *(const v16h*)(w0 + wofs);
      v16h b1 = *(const v16h*)(w0 + (long)16 * Kdim + wofs);
      v16h b2 = *(const v16h*)(w0 + (long)32 * Kdim + wofs);
      v16h b3 = *(const v16h*)(w0 + (long)48 * Kdim + wofs);
      acc[0] = __builtin_amdgcn_wmma_f32_16x16x32_f16(false, af, false, b0, (short)0, acc[0], false, false);
      acc[1] = __builtin_amdgcn_wmma_f32_16x16x32_f16(false, af, false, b1, (short)0, acc[1], false, false);
      acc[2] = __builtin_amdgcn_wmma_f32_16x16x32_f16(false, af, false, b2, (short)0, acc[2], false, false);
      acc[3] = __builtin_amdgcn_wmma_f32_16x16x32_f16(false, af, false, b3, (short)0, acc[3], false, false);
    }
  }

  const int colb = ncol0 + mr;
#pragma unroll
  for (int j = 0; j < 4; ++j) {
    int   col  = colb + j * 16;
    float bcol = bias ? bias[col] : 0.f;
#pragma unroll
    for (int r = 0; r < 8; ++r) {
      int orow = tileM * 16 + half * 8 + r;   // C layout: VGPR r -> row 8*half+r
      if (orow < M) {
        long  idx = (long)orow * ldc + col;
        float v   = acc[j][r] + bcol;
        if (residual) v += residual[idx];
        if (relu && v < 0.f) v = 0.f;
        C[idx] = (TC)v;
      }
    }
  }
}

template <typename TA, typename TC>
static inline void gemmL(hipStream_t st, const TA* A, int lda, const half_t* Wt,
                         const float* bias, const float* residual,
                         TC* C, int ldc, int M, int N, int Kd, int relu)
{
  if (M <= 0) return;
  dim3 g((M + 63) / 64, N / 64), b(128);
  hipLaunchKernelGGL((wmma_gemm_kernel<TA, TC>), g, b, 0, st,
                     A, lda, Wt, bias, residual, C, ldc, M, N, Kd, relu);
}

// Wt[n*Kpad + k] = k < Ksrc ? (f16) W[k*N + n] : 0
__global__ void wt_convert_kernel(const float* __restrict__ W, half_t* __restrict__ Wt,
                                  int Ksrc, int Kpad, int N)
{
  int idx = blockIdx.x * blockDim.x + threadIdx.x;
  if (idx >= Kpad * N) return;
  int n = idx / Kpad, k = idx - n * Kpad;
  Wt[idx] = (k < Ksrc) ? (half_t)W[(long)k * N + n] : (half_t)0.f;
}

// ---------------------------------------------------------------------------
// LayerNorm over 128 features (optionally fused ReLU).  In-place safe.
// ---------------------------------------------------------------------------
__global__ __launch_bounds__(128)
void ln128_kernel(const float* __restrict__ x, const float* __restrict__ g,
                  const float* __restrict__ b, float* __restrict__ y,
                  int M, int relu)
{
  int row = blockIdx.x;
  if (row >= M) return;
  int   c = threadIdx.x;
  float v = x[(long)row * HN + c];
  __shared__ float sred[HN];
  sred[c] = v; __syncthreads();
  for (int s = 64; s > 0; s >>= 1) { if (c < s) sred[c] += sred[c + s]; __syncthreads(); }
  float mean = sred[0] / 128.f; __syncthreads();
  float dv = v - mean;
  sred[c] = dv * dv; __syncthreads();
  for (int s = 64; s > 0; s >>= 1) { if (c < s) sred[c] += sred[c + s]; __syncthreads(); }
  float var = sred[0] / 128.f;
  float o = dv * rsqrtf(var + 1e-5f) * g[c] + b[c];
  if (relu && o < 0.f) o = 0.f;
  y[(long)row * HN + c] = o;
}

// ---------------------------------------------------------------------------
// Agent embedding: MLP(4->128->128) + type_emb + identity_emb   (tiny; VALU)
// ---------------------------------------------------------------------------
__global__ __launch_bounds__(128)
void agent_emb_kernel(const float* __restrict__ a_box,
                      const float* __restrict__ W1, const float* __restrict__ b1,
                      const float* __restrict__ ga, const float* __restrict__ be,
                      const float* __restrict__ W2, const float* __restrict__ b2,
                      const float* __restrict__ type_emb, const float* __restrict__ id_emb,
                      const int* __restrict__ a_type, const int* __restrict__ a_id,
                      float* __restrict__ out)
{
  int row = blockIdx.x, c = threadIdx.x;
  float h = b1[c];
  for (int j = 0; j < 4; ++j) h += a_box[row * 4 + j] * W1[j * HN + c];
  __shared__ float sred[HN];
  __shared__ float sh[HN];
  sred[c] = h; __syncthreads();
  for (int s = 64; s > 0; s >>= 1) { if (c < s) sred[c] += sred[c + s]; __syncthreads(); }
  float mean = sred[0] / 128.f; __syncthreads();
  float dv = h - mean;
  sred[c] = dv * dv; __syncthreads();
  for (int s = 64; s > 0; s >>= 1) { if (c < s) sred[c] += sred[c + s]; __syncthreads(); }
  float var = sred[0] / 128.f;
  float ln = dv * rsqrtf(var + 1e-5f) * ga[c] + be[c];
  sh[c] = ln > 0.f ? ln : 0.f; __syncthreads();
  float o = b2[c];
  for (int j = 0; j < HN; ++j) o += sh[j] * W2[j * HN + c];
  out[(long)row * HN + c] = o + type_emb[a_type[row] * HN + c] + id_emb[a_id[row] * HN + c];
}

// fused_in[K x 256] = [ a_embs[a] | token_tables[type[a], tok[a,t]] ]
__global__ __launch_bounds__(256)
void fuse_build_kernel(const float* __restrict__ a_embs,
                       const float* __restrict__ tok_tables,
                       const int* __restrict__ a_type, const int* __restrict__ recon_token,
                       float* __restrict__ fused)
{
  int k = blockIdx.x, c = threadIdx.x;
  int a = k / TN, t = k % TN;
  float v;
  if (c < HN) v = a_embs[(long)a * HN + c];
  else {
    int ty = a_type[a], tk = recon_token[a * TN + t];
    v = tok_tables[((long)ty * 2048 + tk) * HN + (c - HN)];
  }
  fused[(long)k * 256 + c] = v;
}

// transposed node-space positions: posT[t*A+a] = recon_position[a,t]
__global__ void build_posT_kernel(const float* __restrict__ pos, const float* __restrict__ head,
                                  float* __restrict__ posT, float* __restrict__ headT)
{
  int i = blockIdx.x * blockDim.x + threadIdx.x;
  if (i >= KN) return;
  int a = i % AN, t = i / AN;
  posT[2 * i]     = pos[(a * TN + t) * 2];
  posT[2 * i + 1] = pos[(a * TN + t) * 2 + 1];
  headT[i]        = head[a * TN + t];
}

// edge features (length, cos/sin theta, cos/sin dheading [, extra]);
// row stride FPAD, cols >= 6 stay zero (matched by zero-padded weights)
__global__ void edge_feat_kernel(const float* __restrict__ sp, const float* __restrict__ sh,
                                 const float* __restrict__ dp, const float* __restrict__ dh,
                                 const int* __restrict__ ei, int E, int mode,
                                 const float* __restrict__ extra,
                                 float* __restrict__ F)
{
  int e = blockIdx.x * blockDim.x + threadIdx.x;
  if (e >= E) return;
  int s = ei[e], d = ei[E + e];
  float hs = sh[s], hd = dh[d];
  float cc = cosf(hd), ss = sinf(hd);
  float dx = sp[2 * s] - dp[2 * d], dy = sp[2 * s + 1] - dp[2 * d + 1];
  float lx = cc * dx + ss * dy, ly = -ss * dx + cc * dy;
  float len = sqrtf(lx * lx + ly * ly);
  float th  = atan2f(ly, lx);
  const float PI = 3.14159265358979323846f;
  float w = fmodf(hs - hd + PI, 2.f * PI);
  if (w < 0.f) w += 2.f * PI;
  w -= PI;
  float* o = F + (long)e * FPAD;
  o[0] = len; o[1] = cosf(th); o[2] = sinf(th); o[3] = cosf(w); o[4] = sinf(w);
  if (mode == 0)      o[5] = (float)(s - d) * 5.0f;   // interval feature
  else if (mode == 1) o[5] = extra[s];                // heading_valid_mask
  else                o[5] = 0.f;
}

// kf[e,:] += kn[src[e],:]   (f16 target; 32 threads x 4 halves)
__global__ __launch_bounds__(32)
void edge_gather_add_kernel(half_t* __restrict__ kf, const float* __restrict__ kn,
                            const int* __restrict__ ei, int E)
{
  int e = blockIdx.x;
  if (e >= E) return;
  int  s  = ei[e];
  int  c4 = threadIdx.x * 4;
  v4h  kv = *(v4h*)(kf + (long)e * HN + c4);
  float4 nv = *(const float4*)(kn + (long)s * HN + c4);
  kv[0] = (half_t)((float)kv[0] + nv.x);
  kv[1] = (half_t)((float)kv[1] + nv.y);
  kv[2] = (half_t)((float)kv[2] + nv.z);
  kv[3] = (half_t)((float)kv[3] + nv.w);
  *(v4h*)(kf + (long)e * HN + c4) = kv;
}

// logits + segment max (float order-preserving uint encoding); vector loads
__global__ void logit_max_kernel(const float* __restrict__ q, const half_t* __restrict__ kf,
                                 const int* __restrict__ ei, float* __restrict__ logits,
                                 unsigned* __restrict__ mx, int E)
{
  int idx = blockIdx.x * blockDim.x + threadIdx.x;
  if (idx >= E * 8) return;
  int e = idx >> 3, h = idx & 7;
  int d = ei[E + e];
  const float4* qp = (const float4*)(q + (long)d * HN + h * 16);
  const v8h*    kp = (const v8h*)(kf + (long)e * HN + h * 16);
  float4 q0 = qp[0], q1 = qp[1], q2 = qp[2], q3 = qp[3];
  v8h    k0 = kp[0], k1 = kp[1];
  float dot = q0.x * (float)k0[0] + q0.y * (float)k0[1] + q0.z * (float)k0[2] + q0.w * (float)k0[3]
            + q1.x * (float)k0[4] + q1.y * (float)k0[5] + q1.z * (float)k0[6] + q1.w * (float)k0[7]
            + q2.x * (float)k1[0] + q2.y * (float)k1[1] + q2.z * (float)k1[2] + q2.w * (float)k1[3]
            + q3.x * (float)k1[4] + q3.y * (float)k1[5] + q3.z * (float)k1[6] + q3.w * (float)k1[7];
  float lg = dot * 0.25f;                      // 1/sqrt(head_dim=16)
  logits[idx] = lg;
  unsigned bits = __float_as_uint(lg);
  unsigned key  = (bits & 0x80000000u) ? ~bits : (bits | 0x80000000u);
  atomicMax(&mx[d * 8 + h], key);
}

__global__ void exp_den_kernel(float* __restrict__ logits, const unsigned* __restrict__ mx,
                               float* __restrict__ den, const int* __restrict__ ei, int E)
{
  int idx = blockIdx.x * blockDim.x + threadIdx.x;
  if (idx >= E * 8) return;
  int e = idx >> 3, h = idx & 7;
  int d = ei[E + e];
  unsigned key  = mx[d * 8 + h];
  unsigned bits = (key & 0x80000000u) ? (key ^ 0x80000000u) : ~key;
  float m = (((bits >> 23) & 0xFFu) != 0xFFu) ? __uint_as_float(bits) : 0.f;
  float ev = __expf(logits[idx] - m);
  logits[idx] = ev;
  atomicAdd(&den[d * 8 + h], ev);
}

// per (edge, head): 16 weighted-value atomic adds, vector value loads
__global__ void aggregate_kernel(const float* __restrict__ logits, const float* __restrict__ den,
                                 const half_t* __restrict__ vf, const int* __restrict__ ei,
                                 float* __restrict__ agg, int E)
{
  int idx = blockIdx.x * blockDim.x + threadIdx.x;
  if (idx >= E * 8) return;
  int e = idx >> 3, h = idx & 7;
  int d = ei[E + e];
  float alpha = logits[idx] / (den[d * 8 + h] + 1e-16f);
  const v8h* vp = (const v8h*)(vf + (long)e * HN + h * 16);
  v8h v0 = vp[0], v1 = vp[1];
  float* ap = agg + (long)d * HN + h * 16;
#pragma unroll
  for (int j = 0; j < 8; ++j) atomicAdd(ap + j,     alpha * (float)v0[j]);
#pragma unroll
  for (int j = 0; j < 8; ++j) atomicAdd(ap + 8 + j, alpha * (float)v1[j]);
}

// (i*d1+j) -> (j*d0+i) node permutation of [.,128] rows (32 thr x float4)
__global__ __launch_bounds__(32)
void transpose_nodes_kernel(const float* __restrict__ in, float* __restrict__ out,
                            int d0, int d1)
{
  int k = blockIdx.x, c4 = threadIdx.x * 4;
  int i = k / d1, j = k % d1;
  *(float4*)(out + ((long)(j * d0 + i)) * HN + c4) =
      *(const float4*)(in + (long)k * HN + c4);
}

// ---------------------------------------------------------------------------
// Host-side orchestration
// ---------------------------------------------------------------------------
// Input flattening assumption: top-level dict in insertion order; nested
// "params" pytree flattened JAX-style (dict keys sorted, lists in order).
enum {
  P_AGENT = 0,       // agent_emb MLP: W1,W2,b1,b2,be,g
  P_FUSION = 6,      // fusion MLP
  P_G2K_ATTN = 12,   // 3 x 20 attn leaves
  P_G2K_EMB = 72,
  P_IDENT = 78,
  P_A_ATTN = 79,     // k2k_a_attn
  P_A_EMB = 139,
  P_T_ATTN = 145,    // k2k_t_attn
  P_T_EMB = 205,
  P_TOKTAB = 211,
  P_TYPE = 212,
  IN_GEMBS = 213, IN_ABOX = 214, IN_ATYPE = 215, IN_AID = 216, IN_TOK = 217,
  IN_RPOS = 218, IN_RHEAD = 219, IN_PPOS = 220, IN_PHEAD = 221, IN_HVM = 222,
  IN_EI_T = 223, IN_EI_G = 224, IN_EI_A = 225,
  N_IN_EXPECTED = 226
};
// attn leaves sorted: Wf1,Wf2,Wk,Wke,Wo,Wq,Wv,Wve,bf1,bf2,bk,bke,bo,bq,bv,bve,n1b,n1g,n2b,n2g
enum { iWf1, iWf2, iWk, iWke, iWo, iWq, iWv, iWve, ibf1, ibf2, ibk, ibke,
       ibo, ibq, ibv, ibve, in1b, in1g, in2b, in2g };

struct AttnW { const half_t *wq, *wk, *wv, *wke, *wve, *wo, *wf1, *wf2; };

static void run_attn(hipStream_t st, const float* const* P, const AttnW& W, float* x,
                     const float* src_raw, int Ns, bool src_is_dst,
                     const int* ei, int E, const half_t* ea,
                     float* xd, float* xs, float* q, float* kn, float* vn,
                     half_t* kf, half_t* vf, float* logits, unsigned* mx,
                     float* den, float* agg, float* xt, float* hb, float* f1)
{
  hipLaunchKernelGGL(ln128_kernel, dim3(KN), dim3(128), 0, st, x, P[in1g], P[in1b], xd, KN, 0);
  const float* xsp = xd;
  if (!src_is_dst) {
    hipLaunchKernelGGL(ln128_kernel, dim3(Ns), dim3(128), 0, st,
                       src_raw, P[in1g], P[in1b], xs, Ns, 0);
    xsp = xs;
  }
  gemmL(st, xd,  HN, W.wq, P[ibq], (const float*)nullptr, q,  HN, KN, HN, HN, 0);
  gemmL(st, xsp, HN, W.wk, P[ibk], (const float*)nullptr, kn, HN, Ns, HN, HN, 0);
  gemmL(st, xsp, HN, W.wv, P[ibv], (const float*)nullptr, vn, HN, Ns, HN, HN, 0);
  hipMemsetAsync(mx,  0, (size_t)KN * 8 * sizeof(unsigned), st);
  hipMemsetAsync(den, 0, (size_t)KN * 8 * sizeof(float), st);
  hipMemsetAsync(agg, 0, (size_t)KN * HN * sizeof(float), st);
  if (E > 0) {
    gemmL(st, ea, HN, W.wke, P[ibke], (const float*)nullptr, kf, HN, E, HN, HN, 0);
    gemmL(st, ea, HN, W.wve, P[ibve], (const float*)nullptr, vf, HN, E, HN, HN, 0);
    hipLaunchKernelGGL(edge_gather_add_kernel, dim3(E), dim3(32), 0, st, kf, kn, ei, E);
    hipLaunchKernelGGL(edge_gather_add_kernel, dim3(E), dim3(32), 0, st, vf, vn, ei, E);
    int thr = E * 8;
    hipLaunchKernelGGL(logit_max_kernel, dim3((thr + 255) / 256), dim3(256), 0, st,
                       q, kf, ei, logits, mx, E);
    hipLaunchKernelGGL(exp_den_kernel, dim3((thr + 255) / 256), dim3(256), 0, st,
                       logits, mx, den, ei, E);
    hipLaunchKernelGGL(aggregate_kernel, dim3((thr + 255) / 256), dim3(256), 0, st,
                       logits, den, vf, ei, agg, E);
  }
  // x' = x + agg@Wo + bo ; then x'' = x' + relu(LN(x')@Wf1+bf1)@Wf2 + bf2
  gemmL(st, agg, HN, W.wo, P[ibo], (const float*)x, xt, HN, KN, HN, HN, 0);
  hipLaunchKernelGGL(ln128_kernel, dim3(KN), dim3(128), 0, st, xt, P[in2g], P[in2b], hb, KN, 0);
  gemmL(st, hb, HN, W.wf1, P[ibf1], (const float*)nullptr, f1, 4 * HN, KN, 4 * HN, HN, 1);
  gemmL(st, f1, 4 * HN, W.wf2, P[ibf2], (const float*)xt, x, HN, KN, HN, 4 * HN, 0);
}

extern "C" void kernel_launch(void* const* d_in, const int* in_sizes, int n_in,
                              void* d_out, int out_size, void* d_ws, size_t ws_size,
                              hipStream_t stream)
{
  (void)out_size; (void)ws_size;
  if (n_in < N_IN_EXPECTED) return;
  auto f32 = [&](int i) { return (const float*)d_in[i]; };
  auto i32 = [&](int i) { return (const int*)d_in[i]; };

  const int Et = in_sizes[IN_EI_T] / 2;
  const int Eg = in_sizes[IN_EI_G] / 2;
  const int Ea = in_sizes[IN_EI_A] / 2;
  long Emax = Et; if (Eg > Emax) Emax = Eg; if (Ea > Emax) Emax = Ea; if (Emax < 1) Emax = 1;

  // ---- dynamic workspace partition ----
  char* p = (char*)d_ws;
  auto alloc = [&](size_t nbytes) { char* r = p; p += (nbytes + 255) & ~(size_t)255; return r; };

  // pre-transposed f16 weights, K zero-padded to a multiple of 32
  auto prepW = [&](int leaf, int Ksrc, int Kpad, int N) -> const half_t* {
    half_t* dst = (half_t*)alloc((size_t)Kpad * N * sizeof(half_t));
    int n = Kpad * N;
    hipLaunchKernelGGL(wt_convert_kernel, dim3((n + 255) / 256), dim3(256), 0, stream,
                       f32(leaf), dst, Ksrc, Kpad, N);
    return dst;
  };

  const half_t* fus1t = prepW(P_FUSION + 0, 256, 256, HN);
  const half_t* fus2t = prepW(P_FUSION + 1, HN, HN, HN);
  const half_t* et1t  = prepW(P_T_EMB + 0, 6, FPAD, HN);
  const half_t* et2t  = prepW(P_T_EMB + 1, HN, HN, HN);
  const half_t* eg1t  = prepW(P_G2K_EMB + 0, 6, FPAD, HN);
  const half_t* eg2t  = prepW(P_G2K_EMB + 1, HN, HN, HN);
  const half_t* ea1t  = prepW(P_A_EMB + 0, 5, FPAD, HN);
  const half_t* ea2t  = prepW(P_A_EMB + 1, HN, HN, HN);
  AttnW wT[3], wG[3], wA[3];
  for (int l = 0; l < 3; ++l) {
    int bt = P_T_ATTN + l * 20, bg = P_G2K_ATTN + l * 20, ba = P_A_ATTN + l * 20;
    AttnW* dsts[3] = { &wT[l], &wG[l], &wA[l] };
    int    bases[3] = { bt, bg, ba };
    for (int s = 0; s < 3; ++s) {
      int b = bases[s];
      dsts[s]->wq  = prepW(b + iWq,  HN, HN, HN);
      dsts[s]->wk  = prepW(b + iWk,  HN, HN, HN);
      dsts[s]->wv  = prepW(b + iWv,  HN, HN, HN);
      dsts[s]->wke = prepW(b + iWke, HN, HN, HN);
      dsts[s]->wve = prepW(b + iWve, HN, HN, HN);
      dsts[s]->wo  = prepW(b + iWo,  HN, HN, HN);
      dsts[s]->wf1 = prepW(b + iWf1, HN, HN, 4 * HN);
      dsts[s]->wf2 = prepW(b + iWf2, 4 * HN, 4 * HN, HN);
    }
  }

  float*  a_embs = (float*)alloc((size_t)AN * HN * 4);
  float*  fused  = (float*)alloc((size_t)KN * 256 * 4);
  float*  kx     = (float*)alloc((size_t)KN * HN * 4);
  float*  kT     = (float*)alloc((size_t)KN * HN * 4);
  float*  posT   = (float*)alloc((size_t)KN * 2 * 4);
  float*  headT  = (float*)alloc((size_t)KN * 4);
  float*  feat   = (float*)alloc((size_t)Emax * FPAD * 4);
  half_t* ea_t   = (half_t*)alloc((size_t)(Et > 0 ? Et : 1) * HN * 2);
  half_t* ea_g   = (half_t*)alloc((size_t)(Eg > 0 ? Eg : 1) * HN * 2);
  half_t* ea_a   = (half_t*)alloc((size_t)(Ea > 0 ? Ea : 1) * HN * 2);
  size_t bigsz = (size_t)Emax * HN * 4;                            // edge-MLP hidden
  if (bigsz < (size_t)KN * 512 * 4) bigsz = (size_t)KN * 512 * 4;  // FFN hidden
  float*    big    = (float*)alloc(bigsz);
  float*    xd     = (float*)alloc((size_t)KN * HN * 4);
  float*    xs     = (float*)alloc((size_t)GN * HN * 4);
  float*    q      = (float*)alloc((size_t)KN * HN * 4);
  float*    kn     = (float*)alloc((size_t)KN * HN * 4);
  float*    vn     = (float*)alloc((size_t)KN * HN * 4);
  half_t*   kf     = (half_t*)alloc((size_t)Emax * HN * 2);
  half_t*   vf     = (half_t*)alloc((size_t)Emax * HN * 2);
  float*    logits = (float*)alloc((size_t)Emax * 8 * 4);
  unsigned* mx     = (unsigned*)alloc((size_t)KN * 8 * 4);
  float*    den    = (float*)alloc((size_t)KN * 8 * 4);
  float*    agg    = (float*)alloc((size_t)KN * HN * 4);
  float*    xt     = (float*)alloc((size_t)KN * HN * 4);
  float*    hb     = (float*)alloc((size_t)KN * HN * 4);

  // zero the padded feature columns once per launch
  hipMemsetAsync(feat, 0, (size_t)Emax * FPAD * 4, stream);

  const float* rpos  = f32(IN_RPOS);   // K x 2 (a-major)
  const float* rhead = f32(IN_RHEAD);  // K     (a-major)

  // ---- node embeddings ----
  hipLaunchKernelGGL(agent_emb_kernel, dim3(AN), dim3(128), 0, stream,
                     f32(IN_ABOX), f32(P_AGENT + 0), f32(P_AGENT + 2),
                     f32(P_AGENT + 5), f32(P_AGENT + 4),
                     f32(P_AGENT + 1), f32(P_AGENT + 3),
                     f32(P_TYPE), f32(P_IDENT),
                     i32(IN_ATYPE), i32(IN_AID), a_embs);
  hipLaunchKernelGGL(fuse_build_kernel, dim3(KN), dim3(256), 0, stream,
                     a_embs, f32(P_TOKTAB), i32(IN_ATYPE), i32(IN_TOK), fused);
  gemmL(stream, fused, 256, fus1t, f32(P_FUSION + 2),
        (const float*)nullptr, xt, HN, KN, HN, 256, 0);
  hipLaunchKernelGGL(ln128_kernel, dim3(KN), dim3(128), 0, stream,
                     xt, f32(P_FUSION + 5), f32(P_FUSION + 4), xt, KN, 1);
  gemmL(stream, xt, HN, fus2t, f32(P_FUSION + 3),
        (const float*)nullptr, kx, HN, KN, HN, HN, 0);

  hipLaunchKernelGGL(build_posT_kernel, dim3((KN + 127) / 128), dim3(128), 0, stream,
                     rpos, rhead, posT, headT);

  // ---- edge-attribute MLPs (features -> 128, via WMMA; K padded to 32) ----
  const int* ei_t = i32(IN_EI_T);
  const int* ei_g = i32(IN_EI_G);
  const int* ei_a = i32(IN_EI_A);
  if (Et > 0) {
    hipLaunchKernelGGL(edge_feat_kernel, dim3((Et + 127) / 128), dim3(128), 0, stream,
                       rpos, rhead, rpos, rhead, ei_t, Et, 0, (const float*)nullptr, feat);
    gemmL(stream, feat, FPAD, et1t, f32(P_T_EMB + 2),
          (const float*)nullptr, big, HN, Et, HN, FPAD, 0);
    hipLaunchKernelGGL(ln128_kernel, dim3(Et), dim3(128), 0, stream,
                       big, f32(P_T_EMB + 5), f32(P_T_EMB + 4), big, Et, 1);
    gemmL(stream, big, HN, et2t, f32(P_T_EMB + 3),
          (const float*)nullptr, ea_t, HN, Et, HN, HN, 0);
  }
  if (Eg > 0) {
    hipLaunchKernelGGL(edge_feat_kernel, dim3((Eg + 127) / 128), dim3(128), 0, stream,
                       f32(IN_PPOS), f32(IN_PHEAD), rpos, rhead, ei_g, Eg, 1,
                       f32(IN_HVM), feat);
    gemmL(stream, feat, FPAD, eg1t, f32(P_G2K_EMB + 2),
          (const float*)nullptr, big, HN, Eg, HN, FPAD, 0);
    hipLaunchKernelGGL(ln128_kernel, dim3(Eg), dim3(128), 0, stream,
                       big, f32(P_G2K_EMB + 5), f32(P_G2K_EMB + 4), big, Eg, 1);
    gemmL(stream, big, HN, eg2t, f32(P_G2K_EMB + 3),
          (const float*)nullptr, ea_g, HN, Eg, HN, HN, 0);
  }
  if (Ea > 0) {
    hipLaunchKernelGGL(edge_feat_kernel, dim3((Ea + 127) / 128), dim3(128), 0, stream,
                       posT, headT, posT, headT, ei_a, Ea, 2, (const float*)nullptr, feat);
    gemmL(stream, feat, FPAD, ea1t, f32(P_A_EMB + 2),
          (const float*)nullptr, big, HN, Ea, HN, FPAD, 0);
    hipLaunchKernelGGL(ln128_kernel, dim3(Ea), dim3(128), 0, stream,
                       big, f32(P_A_EMB + 5), f32(P_A_EMB + 4), big, Ea, 1);
    gemmL(stream, big, HN, ea2t, f32(P_A_EMB + 3),
          (const float*)nullptr, ea_a, HN, Ea, HN, HN, 0);
  }

  // ---- 3 layers x (k2k_t, g2k, k2k_a) graph attention ----
  for (int l = 0; l < 3; ++l) {
    const float* Pt[20]; const float* Pg[20]; const float* Pa[20];
    for (int j = 0; j < 20; ++j) {
      Pt[j] = f32(P_T_ATTN + l * 20 + j);
      Pg[j] = f32(P_G2K_ATTN + l * 20 + j);
      Pa[j] = f32(P_A_ATTN + l * 20 + j);
    }
    run_attn(stream, Pt, wT[l], kx, nullptr, KN, true, ei_t, Et, ea_t,
             xd, xs, q, kn, vn, kf, vf, logits, mx, den, agg, xt, hb, big);
    run_attn(stream, Pg, wG[l], kx, f32(IN_GEMBS), GN, false, ei_g, Eg, ea_g,
             xd, xs, q, kn, vn, kf, vf, logits, mx, den, agg, xt, hb, big);
    hipLaunchKernelGGL(transpose_nodes_kernel, dim3(KN), dim3(32), 0, stream, kx, kT, AN, TN);
    run_attn(stream, Pa, wA[l], kT, nullptr, KN, true, ei_a, Ea, ea_a,
             xd, xs, q, kn, vn, kf, vf, logits, mx, den, agg, xt, hb, big);
    hipLaunchKernelGGL(transpose_nodes_kernel, dim3(KN), dim3(32), 0, stream, kT, kx, TN, AN);
  }

  hipMemcpyAsync(d_out, kx, (size_t)KN * HN * sizeof(float),
                 hipMemcpyDeviceToDevice, stream);
}
